// MultiScalePeriodDetector_27144193311246
// MI455X (gfx1250) — matrix operand order repaired and checked by reading, not compiled, not run on hardware
//
#include <hip/hip_runtime.h>
#include <math.h>

// ---------------- problem constants ----------------
#define B_   16
#define T_   2048
#define C_   64
#define N_   1024      // B_*C_
#define F1_  1025      // T/2+1
#define TC_  1027      // (T+6)/2
#define FC_  514       // TC/2+1
#define ALPHA_ 2000.0f
#define TAU_   1e-7f
#define TOL_   1e-6f
#define PI_    3.14159265358979323846f

typedef __attribute__((ext_vector_type(16))) _Float16 v16h;
typedef __attribute__((ext_vector_type(8)))  float    v8f;

__constant__ float DB4[8] = {
    -0.010597401784997278f, 0.032883011666982945f, 0.030841381835986965f,
    -0.18703481171888114f, -0.02798376941698385f, 0.6308807679295904f,
    0.7148465705525415f, 0.23037781330885523f };

// ---------------- workspace layout ----------------
constexpr size_t alignup(size_t v){ return (v + 255) & ~(size_t)255; }
constexpr size_t SZ_X    = (size_t)N_*F1_*sizeof(float);
constexpr size_t OFF_XRE = 0;
constexpr size_t OFF_XIM = alignup(OFF_XRE + SZ_X);
constexpr size_t OFF_CA  = alignup(OFF_XIM + SZ_X);
constexpr size_t OFF_AMPC= alignup(OFF_CA  + (size_t)N_*TC_*sizeof(float));
constexpr size_t OFF_AMPB= alignup(OFF_AMPC+ (size_t)N_*FC_*sizeof(float));
constexpr size_t OFF_MISC= alignup(OFF_AMPB+ (size_t)B_*F1_*sizeof(float)); // 4 ints + 32 floats
constexpr size_t OFF_WIDX= alignup(OFF_MISC + 1024);
constexpr size_t OFF_WEN = alignup(OFF_WIDX + (size_t)N_*2*sizeof(int));
constexpr size_t OFF_VEN = alignup(OFF_WEN  + (size_t)N_*sizeof(float));
constexpr size_t OFF_VFI = alignup(OFF_VEN  + (size_t)N_*2*sizeof(float));

// ---------------- 1) db4 DWT (approximation coeffs, symmetric pad) -----------
__global__ void dwt_db4_kernel(const float* __restrict__ x, float* __restrict__ cA){
  int gid = blockIdx.x*blockDim.x + threadIdx.x;
  if (gid >= N_*TC_) return;
  int n = gid / TC_, j = gid % TC_;
  int b = n >> 6, c = n & 63;
  float acc = 0.f;
  #pragma unroll
  for (int i = 0; i < 8; ++i){
    int q = 2*j + i - 6;                 // ext index minus pad(7), taken at y[2j+1]
    q = (q < 0) ? (-q - 1) : q;          // symmetric reflect
    q = (q >= T_) ? (2*T_ - 1 - q) : q;
    acc += x[((size_t)b*T_ + q)*C_ + c] * DB4[7 - i];
  }
  cA[(size_t)n*TC_ + j] = acc;
}

// ---------------- WMMA fragment maps (wave32, 16x16x32 f16) ------------------
// A (16x32): row = lane&15; element e -> K = e + (e>=8?8:0) + (lane>=16?8:0)
//            => both 8-element runs are K-contiguous (offsets hi*8 and 16+hi*8)
// B (32x16): col = lane&15; element e -> K = e + (e>=8?8:0) + (lane>=16?8:0), same as A
// C/D (16x16 f32): element i -> row = i + (lane>=16?8:0), col = lane&15

// ---------------- 2a) spectrum of x rows via DFT-GEMM on WMMA ----------------
// A-panel staged to LDS through the CDNA5 async global->LDS copy path.
__global__ void __launch_bounds__(256)
dft_x_kernel(const float* __restrict__ x, float* __restrict__ Xre, float* __restrict__ Xim){
  __shared__ __align__(16) float    aTile[128*32];   // [row_local][t_local], 16 KB
  __shared__ __align__(32) _Float16 bfr[32*16];      // B cos, fragment order [lane][e]
  __shared__ __align__(32) _Float16 bfi[32*16];      // B sin, fragment order
  const int tid  = threadIdx.x;
  const int wave = tid >> 5, lane = tid & 31;
  const int f0   = blockIdx.x * 16;
  const int r0   = blockIdx.y * 128 + wave * 16;
  const int m    = lane & 15;
  const int hi   = (lane >= 16);
  const int rl   = wave*16 + m;
  v8f accre = {}; v8f accim = {};
  for (int kb = 0; kb < T_; kb += 32){
    // --- async-stage the 128x32 A tile (t-major => coalesced global bursts) ---
    #pragma unroll
    for (int s2 = 0; s2 < 16; ++s2){
      int idx = tid + s2*256;            // 0..4095
      int tl  = idx >> 7;                // t_local 0..31
      int rr  = idx & 127;               // row_local 0..127
      int row = blockIdx.y*128 + rr;
      int t   = kb + tl;
      unsigned long long ga = (unsigned long long)(uintptr_t)
          &x[((size_t)(row >> 6)*T_ + t)*C_ + (row & 63)];
      // flat LDS address truncated to 32 bits == DS byte offset (ISA 10.2)
      unsigned lo = (unsigned)(uintptr_t)&aTile[rr*32 + tl];
      asm volatile("global_load_async_to_lds_b32 %0, %1, off"
                   :: "v"(lo), "v"(ga) : "memory");
    }
    // --- trig B panel, written directly in fragment order ---
    #pragma unroll
    for (int u = 0; u < 2; ++u){
      int idx = tid*2 + u;               // 0..511
      int kk = idx >> 4, nn = idx & 15;
      int t  = kb + kk;
      int f  = f0 + nn;
      int mm = (f * t) % T_;             // exact integer angle reduction
      float th = -2.f * PI_ * (float)mm * (1.0f/(float)T_);
      float s, c2; __sincosf(th, &s, &c2);
      int pos = (nn + ((kk >> 3) & 1)*16)*16 + (kk & 7) + ((kk >= 16) ? 8 : 0);
      bfr[pos] = (_Float16)c2;
      bfi[pos] = (_Float16)s;
    }
    asm volatile("s_wait_asynccnt 0x0" ::: "memory");
    __syncthreads();
    // --- build fragments (vectorized ds reads) ---
    const float* ap = &aTile[rl*32 + hi*8];
    float4 a0 = *(const float4*)(ap);
    float4 a1 = *(const float4*)(ap + 4);
    float4 a2 = *(const float4*)(ap + 16);
    float4 a3 = *(const float4*)(ap + 20);
    v16h a;
    a[0]=(_Float16)a0.x; a[1]=(_Float16)a0.y; a[2]=(_Float16)a0.z; a[3]=(_Float16)a0.w;
    a[4]=(_Float16)a1.x; a[5]=(_Float16)a1.y; a[6]=(_Float16)a1.z; a[7]=(_Float16)a1.w;
    a[8]=(_Float16)a2.x; a[9]=(_Float16)a2.y; a[10]=(_Float16)a2.z; a[11]=(_Float16)a2.w;
    a[12]=(_Float16)a3.x; a[13]=(_Float16)a3.y; a[14]=(_Float16)a3.z; a[15]=(_Float16)a3.w;
    v16h br = *(const v16h*)&bfr[lane*16];
    v16h bi = *(const v16h*)&bfi[lane*16];
    accre = __builtin_amdgcn_wmma_f32_16x16x32_f16(false, a, false, br, (short)0, accre, false, false);
    accim = __builtin_amdgcn_wmma_f32_16x16x32_f16(false, a, false, bi, (short)0, accim, false, false);
    __syncthreads();
  }
  const int fcol = f0 + m;
  if (fcol < F1_){
    #pragma unroll
    for (int i = 0; i < 8; ++i){
      int rr = r0 + i + (hi ? 8 : 0);
      Xre[(size_t)rr*F1_ + fcol] = accre[i];
      Xim[(size_t)rr*F1_ + fcol] = accim[i];
    }
  }
}

// ---------------- 2b) spectrum amplitude of cA rows (length 1027) ------------
__global__ void __launch_bounds__(256)
dft_ca_kernel(const float* __restrict__ cA, float* __restrict__ ampc){
  __shared__ __align__(32) _Float16 bfr[32*16];
  __shared__ __align__(32) _Float16 bfi[32*16];
  const int tid  = threadIdx.x;
  const int wave = tid >> 5, lane = tid & 31;
  const int f0   = blockIdx.x * 16;
  const int r0   = blockIdx.y * 128 + wave * 16;
  const int m    = lane & 15;
  const int row  = r0 + m;
  const int hi   = (lane >= 16);
  v8f accre = {}; v8f accim = {};
  for (int kb = 0; kb < 1056; kb += 32){      // 33 K-steps, zero-padded past 1027
    #pragma unroll
    for (int u = 0; u < 2; ++u){
      int idx = tid*2 + u;
      int kk = idx >> 4, nn = idx & 15;
      int t  = kb + kk;
      int f  = f0 + nn;
      int mm = (f * t) % TC_;
      float th = -2.f * PI_ * (float)mm * (1.0f/(float)TC_);
      float s, c2; __sincosf(th, &s, &c2);
      int pos = (nn + ((kk >> 3) & 1)*16)*16 + (kk & 7) + ((kk >= 16) ? 8 : 0);
      bfr[pos] = (_Float16)c2;
      bfi[pos] = (_Float16)s;
    }
    __syncthreads();
    if (kb + 32 < TC_)
      __builtin_prefetch(&cA[(size_t)row*TC_ + kb + 32], 0, 1);
    v16h a;
    #pragma unroll
    for (int e = 0; e < 16; ++e){
      int ka = e + ((e >= 8) ? 8 : 0) + (hi ? 8 : 0);
      int t  = kb + ka;
      a[e] = (t < TC_) ? (_Float16)cA[(size_t)row*TC_ + t] : (_Float16)0.0f;
    }
    v16h br = *(const v16h*)&bfr[lane*16];
    v16h bi = *(const v16h*)&bfi[lane*16];
    accre = __builtin_amdgcn_wmma_f32_16x16x32_f16(false, a, false, br, (short)0, accre, false, false);
    accim = __builtin_amdgcn_wmma_f32_16x16x32_f16(false, a, false, bi, (short)0, accim, false, false);
    __syncthreads();
  }
  const int fcol = f0 + m;
  if (fcol < FC_){
    #pragma unroll
    for (int i = 0; i < 8; ++i){
      int rr = r0 + i + (hi ? 8 : 0);
      float re = accre[i], im = accim[i];
      ampc[(size_t)rr*FC_ + fcol] = (fcol == 0) ? 0.f : sqrtf(re*re + im*im);
    }
  }
}

// ---------------- 3) per-(b,f) amplitude mean over channels ------------------
__global__ void ampB_kernel(const float* __restrict__ Xre, const float* __restrict__ Xim,
                            float* __restrict__ ampB){
  int gid = blockIdx.x*blockDim.x + threadIdx.x;
  if (gid >= B_*F1_) return;
  int b = gid / F1_, f = gid % F1_;
  float s = 0.f;
  for (int c = 0; c < C_; ++c){
    size_t o = (size_t)(b*C_ + c)*F1_ + f;
    s += sqrtf(Xre[o]*Xre[o] + Xim[o]*Xim[o]);
  }
  ampB[gid] = s * (1.0f/C_);
}

// ---------------- 4) flist top-2, fft periods & weights ----------------------
__global__ void __launch_bounds__(256)
flist_top2_kernel(const float* __restrict__ ampB, int* __restrict__ mi, float* __restrict__ mf){
  __shared__ float v1[256], v2[256]; __shared__ int j1[256], j2[256];
  float best = -1.f, sec = -1.f; int bi_ = -1, si_ = -1;
  for (int f = threadIdx.x; f < F1_; f += 256){
    float v = 0.f;
    if (f != 0){ for (int b = 0; b < B_; ++b) v += ampB[b*F1_ + f]; v *= (1.0f/B_); }
    if (v > best || (v == best && f > bi_)){ sec = best; si_ = bi_; best = v; bi_ = f; }
    else if (v > sec || (v == sec && f > si_)){ sec = v; si_ = f; }
  }
  v1[threadIdx.x] = best; j1[threadIdx.x] = bi_;
  v2[threadIdx.x] = sec;  j2[threadIdx.x] = si_;
  __syncthreads();
  if (threadIdx.x == 0){
    float B1 = -1.f, B2 = -1.f; int I1 = -1, I2 = -1;
    for (int t = 0; t < 256; ++t){
      float cv[2] = { v1[t], v2[t] }; int ci[2] = { j1[t], j2[t] };
      for (int u = 0; u < 2; ++u){
        float v = cv[u]; int f = ci[u]; if (f < 0) continue;
        if (v > B1 || (v == B1 && f > I1)){ B2 = B1; I2 = I1; B1 = v; I1 = f; }
        else if (v > B2 || (v == B2 && f > I2)){ B2 = v; I2 = f; }
      }
    }
    if (I1 < 1) I1 = 1; if (I2 < 1) I2 = 1;
    mi[0] = I1; mi[1] = I2;
    mi[2] = T_ / I1; mi[3] = T_ / I2;
    for (int b = 0; b < B_; ++b){
      mf[b*2+0] = ampB[b*F1_ + I1];
      mf[b*2+1] = ampB[b*F1_ + I2];
    }
  }
}

// ---------------- 5) wavelet: energy + top-2 bins per sequence ---------------
__global__ void __launch_bounds__(256)
wavelet_topk_kernel(const float* __restrict__ cA, const float* __restrict__ ampc,
                    int* __restrict__ widx, float* __restrict__ wen){
  const int n = blockIdx.x;
  __shared__ float s[256];
  float e = 0.f;
  for (int t = threadIdx.x; t < TC_; t += 256){ float v = cA[(size_t)n*TC_ + t]; e += v*v; }
  s[threadIdx.x] = e; __syncthreads();
  for (int st = 128; st; st >>= 1){ if (threadIdx.x < st) s[threadIdx.x] += s[threadIdx.x+st]; __syncthreads(); }
  if (threadIdx.x == 0) wen[n] = s[0];
  // top-2 (descending, ties -> smaller index, lax.top_k style)
  float best = -1.f, sec = -1.f; int bi_ = FC_, si_ = FC_;
  for (int f = threadIdx.x; f < FC_; f += 256){
    float v = ampc[(size_t)n*FC_ + f];
    if (v > best || (v == best && f < bi_)){ sec = best; si_ = bi_; best = v; bi_ = f; }
    else if (v > sec || (v == sec && f < si_)){ sec = v; si_ = f; }
  }
  __shared__ float fb[256], fs[256]; __shared__ int ib[256], ic[256];
  fb[threadIdx.x] = best; ib[threadIdx.x] = bi_;
  fs[threadIdx.x] = sec;  ic[threadIdx.x] = si_;
  __syncthreads();
  if (threadIdx.x == 0){
    float B1 = -1.f, B2 = -1.f; int I1 = FC_, I2 = FC_;
    for (int t = 0; t < 256; ++t){
      float cv[2] = { fb[t], fs[t] }; int ci2[2] = { ib[t], ic[t] };
      for (int u = 0; u < 2; ++u){
        float v = cv[u]; int f = ci2[u]; if (f >= FC_) continue;
        if (v > B1 || (v == B1 && f < I1)){ B2 = B1; I2 = I1; B1 = v; I1 = f; }
        else if (v > B2 || (v == B2 && f < I2)){ B2 = v; I2 = f; }
      }
    }
    widx[n*2+0] = I1; widx[n*2+1] = I2;
  }
}

// ---------------- 6) VMD: 30 iterations, all state in registers --------------
__device__ __forceinline__ float wang_u01(unsigned s){
  s = (s ^ 61u) ^ (s >> 16); s *= 9u; s ^= s >> 4; s *= 0x27d4eb2du; s ^= s >> 15;
  return (float)(s & 0x00FFFFFFu) * (1.0f/16777216.0f);
}

__global__ void __launch_bounds__(256)
vmd_kernel(const float* __restrict__ Xre, const float* __restrict__ Xim,
           float* __restrict__ ven, int* __restrict__ vfidx){
  const int n = blockIdx.x, tid = threadIdx.x;
  __shared__ float red[256*4];
  __shared__ float Vr[T_], Vi[T_];
  float fhr[8], fhi[8], lr[8], li[8];
  float ur[2][8], ui[2][8], upr[2][8], upi[2][8];
  float omega[2];
  // f_hat = fftshift(full spectrum from rfft Hermitian symmetry)
  #pragma unroll
  for (int j = 0; j < 8; ++j){
    int t = tid*8 + j;
    int f = (t + T_/2) & (T_-1);
    float re, im;
    if (f <= T_/2){ re = Xre[(size_t)n*F1_ + f];        im =  Xim[(size_t)n*F1_ + f]; }
    else          { int g = T_-f; re = Xre[(size_t)n*F1_ + g]; im = -Xim[(size_t)n*F1_ + g]; }
    fhr[j] = re; fhi[j] = im; lr[j] = 0.f; li[j] = 0.f;
    for (int k = 0; k < 2; ++k){ ur[k][j]=0.f; ui[k][j]=0.f; upr[k][j]=0.f; upi[k][j]=0.f; }
  }
  omega[0] = 0.5f * wang_u01((unsigned)(n*2 + 0) + 12345u);
  omega[1] = 0.5f * wang_u01((unsigned)(n*2 + 1) + 12345u);
  const float tau2 = TAU_*TAU_;
  for (int it = 0; it < 30; ++it){
    float sfp0=0.f, sp0=0.f, sfp1=0.f, sp1=0.f;
    float unr[2][8], uni[2][8];
    #pragma unroll
    for (int j = 0; j < 8; ++j){
      int t = tid*8 + j;
      float fr = (float)t*(1.0f/T_) - 0.5f;
      float d0 = (fr-omega[0])*(fr-omega[0]);
      float d1 = (fr-omega[1])*(fr-omega[1]);
      float numr = fhr[j] - 0.5f*lr[j];
      float numi = fhi[j] - 0.5f*li[j];
      float w0 = 1.0f/(1.0f + ALPHA_*(d0 + tau2 + d1));   // k=0: self + right nbr
      float w1 = 1.0f/(1.0f + ALPHA_*(d1 + tau2 + d0));   // k=1: self + left nbr
      unr[0][j]=numr*w0; uni[0][j]=numi*w0;
      unr[1][j]=numr*w1; uni[1][j]=numi*w1;
      float p0 = unr[0][j]*unr[0][j] + uni[0][j]*uni[0][j];
      float p1 = unr[1][j]*unr[1][j] + uni[1][j]*uni[1][j];
      sfp0 += fr*p0; sp0 += p0; sfp1 += fr*p1; sp1 += p1;
    }
    red[tid*4+0]=sfp0; red[tid*4+1]=sp0; red[tid*4+2]=sfp1; red[tid*4+3]=sp1;
    __syncthreads();
    for (int st = 128; st; st >>= 1){
      if (tid < st){ for (int q = 0; q < 4; ++q) red[tid*4+q] += red[(tid+st)*4+q]; }
      __syncthreads();
    }
    float om0 = red[0]/(red[1]+1e-8f);
    float om1 = red[2]/(red[3]+1e-8f);
    __syncthreads();
    #pragma unroll
    for (int j = 0; j < 8; ++j){
      lr[j] += TAU_*(unr[0][j]+unr[1][j]-fhr[j]);
      li[j] += TAU_*(uni[0][j]+uni[1][j]-fhi[j]);
    }
    bool brk = false;
    if (it % 5 == 0){
      float sd = 0.f, sp = 0.f;
      for (int k = 0; k < 2; ++k)
        #pragma unroll
        for (int j = 0; j < 8; ++j){
          float dr = unr[k][j]-upr[k][j], di = uni[k][j]-upi[k][j];
          sd += dr*dr + di*di;
          sp += upr[k][j]*upr[k][j] + upi[k][j]*upi[k][j];
        }
      red[tid*4+0]=sd; red[tid*4+1]=sp;
      __syncthreads();
      for (int st = 128; st; st >>= 1){
        if (tid < st){ red[tid*4+0]+=red[(tid+st)*4+0]; red[tid*4+1]+=red[(tid+st)*4+1]; }
        __syncthreads();
      }
      brk = (red[0]/(red[1]+1e-8f) < TOL_);
      __syncthreads();
    }
    for (int k = 0; k < 2; ++k)
      #pragma unroll
      for (int j = 0; j < 8; ++j){ ur[k][j]=unr[k][j]; ui[k][j]=uni[k][j]; upr[k][j]=unr[k][j]; upi[k][j]=uni[k][j]; }
    omega[0]=om0; omega[1]=om1;
    if (brk) break;                 // per-sequence freeze (whole workgroup uniform)
  }
  // outputs in frequency domain: imf = ifft(V).real, V = ifftshift(u_final)
  // fft(imf)[f] = (V[f]+conj(V[T-f]))/2 ; energy via Parseval
  for (int k = 0; k < 2; ++k){
    #pragma unroll
    for (int j = 0; j < 8; ++j){
      int t = tid*8 + j;
      int f = (t + T_/2) & (T_-1);
      Vr[f] = ur[k][j]; Vi[f] = ui[k][j];
    }
    __syncthreads();
    float en = 0.f, bmax = -1.f; int bidx = 0;
    for (int f = tid; f < T_; f += 256){
      int g = (T_ - f) & (T_-1);
      float gr = 0.5f*(Vr[f] + Vr[g]);
      float gi = 0.5f*(Vi[f] - Vi[g]);
      float a2 = gr*gr + gi*gi;
      en += a2;
      if (f >= 1 && f < F1_){
        float amp = sqrtf(a2);
        if (amp > bmax){ bmax = amp; bidx = f; }
      }
    }
    red[tid*4+0] = en; red[tid*4+1] = bmax; red[tid*4+2] = __int_as_float(bidx);
    __syncthreads();
    for (int st = 128; st; st >>= 1){
      if (tid < st){
        red[tid*4+0] += red[(tid+st)*4+0];
        float v2 = red[(tid+st)*4+1]; int i2 = __float_as_int(red[(tid+st)*4+2]);
        float v1 = red[tid*4+1];      int i1 = __float_as_int(red[tid*4+2]);
        if (v2 > v1 || (v2 == v1 && i2 < i1)){ red[tid*4+1] = v2; red[tid*4+2] = __int_as_float(i2); }
      }
      __syncthreads();
    }
    if (tid == 0){
      ven[n*2+k]   = red[0]*(1.0f/T_);
      vfidx[n*2+k] = __float_as_int(red[2]);
    }
    __syncthreads();
  }
}

// ---------------- 7) host-logic fusion, on device ----------------------------
__global__ void __launch_bounds__(256)
fusion_kernel(const int* __restrict__ mi, const float* __restrict__ mf,
              const int* __restrict__ widx, const float* __restrict__ wen,
              const float* __restrict__ ven, const int* __restrict__ vfidx,
              float* __restrict__ out){
  __shared__ int hist[4096];
  __shared__ float acc[2];
  __shared__ int wp[2], wpc; __shared__ float wscal;
  __shared__ int vp[2], vpc; __shared__ float vscal;
  const int tid = threadIdx.x;
  // ---- wavelet candidates ----
  for (int i = tid; i < 4096; i += 256) hist[i] = 0;
  if (tid == 0){ acc[0] = 0.f; acc[1] = 0.f; }
  __syncthreads();
  for (int i = tid; i < N_*2; i += 256){
    int n = i >> 1;
    int idx = widx[i];
    int per = (TC_ / (idx > 0 ? idx : 1)) * 2;
    if (idx > 0 && per > 1 && per < (T_/2)){
      atomicAdd(&hist[per], 1);
      atomicAdd(&acc[0], wen[n]);
      atomicAdd(&acc[1], 1.0f);
    }
  }
  __syncthreads();
  if (tid == 0){
    int c1=0,p1=-1,c2=0,p2=-1,uniq=0;
    for (int p = 0; p < 4096; ++p){
      int c = hist[p]; if (!c) continue; uniq++;
      if (c >= c1){ c2=c1; p2=p1; c1=c; p1=p; }
      else if (c >= c2){ c2=c; p2=p; }
    }
    if (acc[1] == 0.f){ wp[0]=T_/2; wp[1]=T_/2; wpc=2; wscal=0.5f; }
    else { wp[0]=p1; wpc=1; if (uniq >= 2){ wp[1]=p2; wpc=2; } wscal=(acc[0]/acc[1])*0.5f; }
  }
  __syncthreads();
  // ---- vmd candidates ----
  for (int i = tid; i < 4096; i += 256) hist[i] = 0;
  if (tid == 0){ acc[0] = 0.f; acc[1] = 0.f; }
  __syncthreads();
  for (int n = tid; n < N_; n += 256){
    float e0 = ven[n*2+0], e1 = ven[n*2+1];
    int first = (e1 >= e0) ? 1 : 0;            // argsort desc, ties -> later idx first
    bool all0 = (e0 == 0.f && e1 == 0.f);
    for (int u = 0; u < 2; ++u){
      int sel = (u == 0) ? first : (1 - first);
      int f = vfidx[n*2+sel];
      if (!all0 && f > 0){
        int per = T_ / f;
        atomicAdd(&hist[per], 1);
        atomicAdd(&acc[0], ven[n*2+sel]);
        atomicAdd(&acc[1], 1.0f);
      }
    }
  }
  __syncthreads();
  if (tid == 0){
    int c1=0,p1=-1,c2=0,p2=-1,uniq=0,mc=0,mcp=-1;
    for (int p = 0; p < 4096; ++p){
      int c = hist[p]; if (!c) continue; uniq++;
      if (c >= c1){ c2=c1; p2=p1; c1=c; p1=p; }
      else if (c >= c2){ c2=c; p2=p; }
      if (c > mc){ mc=c; mcp=p; }              // np.argmax: first (smallest period)
    }
    if (acc[1] == 0.f){ vp[0]=T_/2; vp[1]=T_/2; vpc=2; vscal=0.5f; }
    else if (uniq < 2){ vp[0]=p1; vp[1]=mcp; vpc=2; vscal=(acc[0]/acc[1])*0.5f; }
    else { vp[0]=p1; vp[1]=p2; vpc=2; vscal=(acc[0]/acc[1])*0.5f; }
  }
  __syncthreads();
  // ---- fuse ----
  if (tid == 0){
    int fftp0 = mi[2], fftp1 = mi[3];
    int allc[6]; int na = 0;
    allc[na++] = fftp0; allc[na++] = fftp1;
    for (int u = 0; u < wpc; ++u) allc[na++] = wp[u];
    for (int u = 0; u < vpc; ++u) allc[na++] = vp[u];
    int vals[6], cnts[6], nu = 0;
    for (int i = 0; i < na; ++i){
      int v = allc[i]; bool seen = false;
      for (int j = 0; j < nu; ++j) if (vals[j] == v){ cnts[j]++; seen = true; break; }
      if (!seen){ vals[nu] = v; cnts[nu] = 1; nu++; }
    }
    for (int i = 1; i < nu; ++i){                 // ascending order (np.unique)
      int v = vals[i], c = cnts[i], j = i-1;
      while (j >= 0 && vals[j] > v){ vals[j+1]=vals[j]; cnts[j+1]=cnts[j]; j--; }
      vals[j+1] = v; cnts[j+1] = c;
    }
    int bc1=0,bp1=-1,bc2=0,bp2=-1,any=0;
    for (int i = 0; i < nu; ++i){
      int v = vals[i]; if (!(v > 1 && v < T_/2)) continue; any++;
      int c = cnts[i];
      if (c >= bc1){ bc2=bc1; bp2=bp1; bc1=c; bp1=v; }
      else if (c >= bc2){ bc2=c; bp2=v; }
    }
    if (any == 0){   // fft_only
      out[0] = (float)fftp0; out[1] = (float)fftp1;
      for (int b = 0; b < B_; ++b){ out[2+b*2+0]=mf[b*2+0]; out[2+b*2+1]=mf[b*2+1]; }
      return;
    }
    int fused[2]; int nf = 0;
    fused[nf++] = bp1; if (any >= 2) fused[nf++] = bp2;
    out[0] = (float)fused[0];
    out[1] = (nf > 1) ? (float)fused[1] : 0.0f;
    for (int col = 0; col < 2; ++col){
      if (col < nf){
        int p = fused[col];
        int jf = (fftp0 == p) ? 0 : ((fftp1 == p) ? 1 : -1);
        bool inw = (p == wp[0]) || (wpc > 1 && p == wp[1]);
        bool inv = (p == vp[0]) || (vpc > 1 && p == vp[1]);
        int cnt = (jf >= 0 ? 1 : 0) + (inw ? 1 : 0) + (inv ? 1 : 0);
        for (int b = 0; b < B_; ++b){
          float s = 0.f;
          if (jf >= 0) s += mf[b*2+jf];
          if (inw) s += wscal;
          if (inv) s += vscal;
          out[2+b*2+col] = (cnt > 0) ? s/(float)cnt : 0.0f;
        }
      } else {
        for (int b = 0; b < B_; ++b) out[2+b*2+col] = 0.0f;
      }
    }
  }
}

// ---------------- launcher ----------------
extern "C" void kernel_launch(void* const* d_in, const int* in_sizes, int n_in,
                              void* d_out, int out_size, void* d_ws, size_t ws_size,
                              hipStream_t stream){
  (void)in_sizes; (void)n_in; (void)out_size; (void)ws_size;
  const float* x = (const float*)d_in[0];
  char* ws = (char*)d_ws;
  float* Xre  = (float*)(ws + OFF_XRE);
  float* Xim  = (float*)(ws + OFF_XIM);
  float* cA   = (float*)(ws + OFF_CA);
  float* ampc = (float*)(ws + OFF_AMPC);
  float* ampB = (float*)(ws + OFF_AMPB);
  int*   mi   = (int*)  (ws + OFF_MISC);
  float* mf   = (float*)(ws + OFF_MISC + 64);
  int*   widx = (int*)  (ws + OFF_WIDX);
  float* wen  = (float*)(ws + OFF_WEN);
  float* ven  = (float*)(ws + OFF_VEN);
  int*   vfi  = (int*)  (ws + OFF_VFI);

  dwt_db4_kernel     <<<(N_*TC_ + 255)/256, 256, 0, stream>>>(x, cA);
  dft_x_kernel       <<<dim3(65, 8), 256, 0, stream>>>(x, Xre, Xim);
  dft_ca_kernel      <<<dim3(33, 8), 256, 0, stream>>>(cA, ampc);
  ampB_kernel        <<<(B_*F1_ + 255)/256, 256, 0, stream>>>(Xre, Xim, ampB);
  flist_top2_kernel  <<<1, 256, 0, stream>>>(ampB, mi, mf);
  wavelet_topk_kernel<<<N_, 256, 0, stream>>>(cA, ampc, widx, wen);
  vmd_kernel         <<<N_, 256, 0, stream>>>(Xre, Xim, ven, vfi);
  fusion_kernel      <<<1, 256, 0, stream>>>(mi, mf, widx, wen, ven, vfi, (float*)d_out);
}
